// ASCPA_39719857553958
// MI455X (gfx1250) — matrix operand order repaired and checked by reading, not compiled, not run on hardware
//
#include <hip/hip_runtime.h>
#include <math.h>

typedef _Float16 h8   __attribute__((ext_vector_type(8)));
typedef _Float16 v16h __attribute__((ext_vector_type(16)));
typedef float    v8f  __attribute__((ext_vector_type(8)));
typedef unsigned int v4u __attribute__((ext_vector_type(4)));
typedef int      v8i  __attribute__((ext_vector_type(8)));
typedef int      v4i  __attribute__((ext_vector_type(4)));

#define C1    256
#define INTER 32
#define NPIX  4096
#define C3    768   // 3 * C1 channels of the fused Gram operand
#define BATCH 2

#if defined(__AMDGCN__) && __has_builtin(__builtin_amdgcn_tensor_load_to_lds) && __has_builtin(__builtin_amdgcn_s_wait_tensorcnt)
#define USE_TDM 1
#else
#define USE_TDM 0
#endif

__device__ __forceinline__ v8f wmma16x16x32(v16h a, v16h b, v8f c) {
#if defined(__AMDGCN__)
  return __builtin_amdgcn_wmma_f32_16x16x32_f16(false, a, false, b, (short)0, c, false, false);
#else
  return c;
#endif
}

__device__ __forceinline__ v16h hcat(h8 lo, h8 hi) {
  return __builtin_shufflevector(lo, hi, 0,1,2,3,4,5,6,7,8,9,10,11,12,13,14,15);
}

#if USE_TDM
__device__ __forceinline__ unsigned lds_off(const void* p) {
  // generic pointer to LDS: low 32 bits are the LDS byte offset
  return (unsigned)(unsigned long long)p;
}
// Build a 2-D Tensor-DMA descriptor (D#) per CDNA5 ISA ch.8 and issue TENSOR_LOAD_TO_LDS.
// data_size = 2 bytes (f16). tile rows land contiguously in LDS (row-major [tile1][tile0]).
// This toolchain's builtin takes 6 args: group0 (v4u), group1 (v8i), group2 (v4i),
// group3 (v4i), an extra group (v8i), and cpol. Groups 2+ are unused for 2-D tensors.
__device__ __forceinline__ void tdm_load_2d(unsigned ldsOff, const void* g,
                                            unsigned td0, unsigned td1,
                                            unsigned tile0, unsigned tile1,
                                            unsigned long long stride0) {
  unsigned long long ga = (unsigned long long)g;
  v4u g0;
  g0[0] = 1u;                                  // count=1, user mode, no gather
  g0[1] = ldsOff;                              // lds_addr (bytes)
  g0[2] = (unsigned)ga;                        // global_addr[31:0]
  g0[3] = (unsigned)(ga >> 32) | 0x80000000u;  // global_addr[56:32] | type=2
  v8i g1;
  g1[0] = (int)(1u << 16);                     // wg_mask=0, data_size=1 (2B)
  g1[1] = (int)((td0 & 0xFFFFu) << 16);                         // tensor_dim0 lo -> bits79:48
  g1[2] = (int)((td0 >> 16) | ((td1 & 0xFFFFu) << 16));         // dim0 hi | dim1 lo
  g1[3] = (int)(((td1 >> 16) & 0xFFFFu) | (tile0 << 16));       // dim1 hi | tile_dim0
  g1[4] = (int)(tile1 & 0xFFFFu);                               // tile_dim1, tile_dim2=0
  g1[5] = (int)(unsigned)(stride0 & 0xFFFFFFFFull);             // tensor_dim0_stride lo
  g1[6] = (int)(unsigned)((stride0 >> 32) & 0xFFFFull);         // stride0 hi | stride1 lo=0
  g1[7] = 0;
  v4i gz4 = {0, 0, 0, 0};
  v8i gz8 = {0, 0, 0, 0, 0, 0, 0, 0};
  __builtin_amdgcn_tensor_load_to_lds(g0, g1, gz4, gz4, gz8, 0);
}
#endif

// ---------------- Kernel 1: 3x3 / 5x5 avg pools + per-channel row sums ----------------
__global__ void __launch_bounds__(256) pool_stats_kernel(const float* __restrict__ x,
    float* __restrict__ X3, float* __restrict__ X5, float* __restrict__ PART) {
  const int b = blockIdx.z, c = blockIdx.y, tblk = blockIdx.x;
  const int tid = threadIdx.x;
  const int n = tblk * 256 + tid;
  const int h = n >> 6, w = n & 63;
  const float* xc = x + ((size_t)(b * C1 + c)) * NPIX;
  const float v0 = xc[n];
  float s3 = 0.f, s5 = 0.f;
  #pragma unroll
  for (int dh = -2; dh <= 2; ++dh) {
    const int hh = h + dh;
    if (hh < 0 || hh > 63) continue;
    #pragma unroll
    for (int dw = -2; dw <= 2; ++dw) {
      const int ww = w + dw;
      if (ww < 0 || ww > 63) continue;
      const float val = xc[hh * 64 + ww];
      s5 += val;
      if (dh >= -1 && dh <= 1 && dw >= -1 && dw <= 1) s3 += val;
    }
  }
  const float p3 = s3 * (1.f / 9.f), p5 = s5 * (1.f / 25.f);
  const size_t o = ((size_t)(b * C1 + c)) * NPIX + n;
  X3[o] = p3; X5[o] = p5;
  __shared__ float red0[256], red1[256], red2[256];
  red0[tid] = v0; red1[tid] = p3; red2[tid] = p5;
  __syncthreads();
  for (int s = 128; s > 0; s >>= 1) {
    if (tid < s) { red0[tid] += red0[tid+s]; red1[tid] += red1[tid+s]; red2[tid] += red2[tid+s]; }
    __syncthreads();
  }
  if (tid == 0) {
    const size_t pb = ((size_t)(b * C1 + c)) * 3;
    PART[(pb + 0) * 16 + tblk] = red0[0];
    PART[(pb + 1) * 16 + tblk] = red1[0];
    PART[(pb + 2) * 16 + tblk] = red2[0];
  }
}

// ---------------- Kernel 2: gate V = softmax(W2 W1 v), v_k = sum_c rowsum_k^2 / N^2 ----
__global__ void __launch_bounds__(256) stats_kernel(const float* __restrict__ PART,
    const float* __restrict__ W1, const float* __restrict__ W2, float* __restrict__ VV) {
  const int b = blockIdx.x, c = threadIdx.x;
  float s0 = 0.f, s1 = 0.f, s2 = 0.f;
  const size_t pb = ((size_t)(b * C1 + c)) * 3;
  for (int t = 0; t < 16; ++t) {
    s0 += PART[(pb + 0) * 16 + t];
    s1 += PART[(pb + 1) * 16 + t];
    s2 += PART[(pb + 2) * 16 + t];
  }
  __shared__ float red0[256], red1[256], red2[256];
  red0[c] = s0 * s0; red1[c] = s1 * s1; red2[c] = s2 * s2;
  __syncthreads();
  for (int st = 128; st > 0; st >>= 1) {
    if (c < st) { red0[c] += red0[c+st]; red1[c] += red1[c+st]; red2[c] += red2[c+st]; }
    __syncthreads();
  }
  if (c == 0) {
    const float inv2 = 1.f / 16777216.f;  // 1/N^2
    float v[3] = { red0[0] * inv2, red1[0] * inv2, red2[0] * inv2 };
    float t16[16];
    for (int j = 0; j < 16; ++j)
      t16[j] = W1[j*3+0]*v[0] + W1[j*3+1]*v[1] + W1[j*3+2]*v[2];
    float u[3];
    for (int k = 0; k < 3; ++k) {
      float a = 0.f;
      for (int j = 0; j < 16; ++j) a += W2[k*16+j] * t16[j];
      u[k] = a;
    }
    const float mx = fmaxf(u[0], fmaxf(u[1], u[2]));
    const float e0 = __expf(u[0]-mx), e1 = __expf(u[1]-mx), e2 = __expf(u[2]-mx);
    const float inv = 1.f / (e0 + e1 + e2);
    VV[b*6+0] = e0*inv; VV[b*6+1] = e1*inv; VV[b*6+2] = e2*inv;
    VV[b*6+3] = sqrtf(e0*inv); VV[b*6+4] = sqrtf(e1*inv); VV[b*6+5] = sqrtf(e2*inv);
  }
}

// ---------------- Kernel 3: XT[b][n][768] = f16( sqrt(V_k) * {x,x3,x5}[c][n] ) ---------
__global__ void __launch_bounds__(256) build_xt_kernel(const float* __restrict__ x,
    const float* __restrict__ X3, const float* __restrict__ X5,
    const float* __restrict__ VV, _Float16* __restrict__ XT) {
  __shared__ _Float16 T[64][72];  // padded vs bank conflicts
  const int b = blockIdx.z;
  const int cb = blockIdx.y;            // 0..11 -> 64-channel block of the 768
  const int n0 = blockIdx.x * 64;
  const int k = cb >> 2;                // source: 0=x, 1=x3, 2=x5
  const int c0 = (cb & 3) * 64;
  const float* src = (k == 0) ? x : ((k == 1) ? X3 : X5);
  const float scale = VV[b*6 + 3 + k];
  const int tid = threadIdx.x;
  const int nl = tid & 63, cg = tid >> 6;
  for (int r = 0; r < 16; ++r) {
    const int cl = cg * 16 + r;
    const float v = src[((size_t)(b * C1 + c0 + cl)) * NPIX + n0 + nl];
    T[cl][nl] = (_Float16)(v * scale);
  }
  __syncthreads();
  const int nl2 = tid >> 2, cs = (tid & 3) * 16;
  _Float16* dst = XT + ((size_t)(b * NPIX) + n0 + nl2) * C3 + cb * 64 + cs;
  for (int j = 0; j < 16; ++j) dst[j] = T[cs + j][nl2];
}

// ---------------- Kernel 3b: GT[b][i][n] = f16( sum_c Wg[i][c] x[b][c][n] ) ------------
__global__ void __launch_bounds__(256) build_g_kernel(const float* __restrict__ x,
    const float* __restrict__ Wg, _Float16* __restrict__ GT) {
  const int b = blockIdx.z, i = blockIdx.y;
  const int n = blockIdx.x * 256 + threadIdx.x;
  const float* xb = x + (size_t)b * C1 * NPIX;
  float acc = 0.f;
  for (int c = 0; c < C1; ++c) acc += Wg[i * C1 + c] * xb[(size_t)c * NPIX + n];
  GT[((size_t)(b * INTER + i)) * NPIX + n] = (_Float16)acc;
}

// ---------------- Kernel 4: flash attention, WMMA f16 -> f32 ---------------------------
// block = 4 waves, each wave owns 16 query rows; loops over all 4096 keys in 32-key
// chunks staged in LDS by the Tensor Data Mover. Scores computed key-major so the
// softmax'd tile packs directly (in-lane) into the A operand of the P*G WMMA.
__global__ void __launch_bounds__(128) flash_kernel(const _Float16* __restrict__ XT,
    const _Float16* __restrict__ GT, float* __restrict__ YW) {
  __shared__ __align__(16) _Float16 Qs[64 * C3];   // 96 KB
  __shared__ __align__(16) _Float16 Ks[32 * C3];   // 48 KB
  __shared__ __align__(16) _Float16 Gs[32 * 32];   //  2 KB

  const int b = blockIdx.y;
  const int q0 = blockIdx.x * 64;
  const int tid = threadIdx.x;
  const int wave = tid >> 5;
  const int lane = tid & 31;
  const int l15 = lane & 15;
  const bool hi = lane >= 16;

  const _Float16* XTb = XT + (size_t)b * NPIX * C3;
  const _Float16* GTb = GT + (size_t)b * INTER * NPIX;

#if USE_TDM
  if (wave == 0) {
    tdm_load_2d(lds_off(Qs), XTb + (size_t)q0 * C3, C3, NPIX, C3, 64, C3);
    __builtin_amdgcn_s_wait_tensorcnt(0);
  }
#else
  for (int idx = tid; idx < 64 * C3 / 8; idx += 128)
    ((uint4*)Qs)[idx] = ((const uint4*)(XTb + (size_t)q0 * C3))[idx];
#endif
  __syncthreads();

  v8f y0 = {}, y1 = {};
  float m_run = -1e30f, l_run = 0.f;
  const int qrow = wave * 16 + l15;

  for (int m0 = 0; m0 < NPIX; m0 += 32) {
#if USE_TDM
    if (wave == 0) {
      tdm_load_2d(lds_off(Ks), XTb + (size_t)m0 * C3, C3, NPIX, C3, 32, C3);
      tdm_load_2d(lds_off(Gs), GTb + m0, NPIX, INTER, 32, 32, NPIX);
      __builtin_amdgcn_s_wait_tensorcnt(0);
    }
#else
    for (int idx = tid; idx < 32 * C3 / 8; idx += 128)
      ((uint4*)Ks)[idx] = ((const uint4*)(XTb + (size_t)m0 * C3))[idx];
    for (int idx = tid; idx < 32 * 32 / 8; idx += 128) {
      const int i = idx >> 2, mseg = idx & 3;
      ((uint4*)Gs)[idx] = *(const uint4*)(GTb + (size_t)i * NPIX + m0 + mseg * 8);
    }
#endif
    __syncthreads();

    // S'[m_local, q_local] over K=768: A = key rows (Ks), B = query cols (Qs)
    v8f Sa = {}, Sb = {};
    for (int kc = 0; kc < 24; ++kc) {
      const int cb = kc * 32;
      const _Float16* qp = &Qs[qrow * C3 + cb + (hi ? 8 : 0)];
      const v16h Bq = hcat(*(const h8*)qp, *(const h8*)(qp + 16));
      const _Float16* ka0 = &Ks[l15 * C3 + cb + (hi ? 8 : 0)];
      const v16h A0 = hcat(*(const h8*)ka0, *(const h8*)(ka0 + 16));
      Sa = wmma16x16x32(A0, Bq, Sa);
      const _Float16* ka1 = &Ks[(16 + l15) * C3 + cb + (hi ? 8 : 0)];
      const v16h A1 = hcat(*(const h8*)ka1, *(const h8*)(ka1 + 16));
      Sb = wmma16x16x32(A1, Bq, Sb);
    }

    // online softmax: lane holds query q=l15; its VGPR r holds key m = r + 8*hi (+16 in Sb)
    float cmax = -1e30f;
    #pragma unroll
    for (int r = 0; r < 8; ++r) cmax = fmaxf(cmax, fmaxf(Sa[r], Sb[r]));
    cmax = fmaxf(cmax, __shfl_xor(cmax, 16, 32));
    const float mnew = fmaxf(m_run, cmax);
    const float corr = __expf(m_run - mnew);
    float pa[8], pb[8];
    float psum = 0.f;
    #pragma unroll
    for (int r = 0; r < 8; ++r) {
      pa[r] = __expf(Sa[r] - mnew);
      pb[r] = __expf(Sb[r] - mnew);
      psum += pa[r] + pb[r];
    }
    psum += __shfl_xor(psum, 16, 32);
    l_run = l_run * corr + psum;
    m_run = mnew;

    // rescale Y accumulators (D rows q = r + 8*hi; factor lives in lane q)
    #pragma unroll
    for (int r = 0; r < 8; ++r) {
      const float cr = __shfl(corr, r + (hi ? 8 : 0), 32);
      y0[r] *= cr; y1[r] *= cr;
    }

    // pack P (16q x 32m) into A operand: in-lane, layouts line up exactly
    v16h Ap;
    #pragma unroll
    for (int r = 0; r < 8; ++r) {
      Ap[r]     = (_Float16)pa[r];
      Ap[8 + r] = (_Float16)pb[r];
    }
    const _Float16* g0p = &Gs[l15 * 32 + (hi ? 8 : 0)];
    const v16h Bg0 = hcat(*(const h8*)g0p, *(const h8*)(g0p + 16));
    const _Float16* g1p = &Gs[(16 + l15) * 32 + (hi ? 8 : 0)];
    const v16h Bg1 = hcat(*(const h8*)g1p, *(const h8*)(g1p + 16));
    y0 = wmma16x16x32(Ap, Bg0, y0);   // y[q, i=0..15]
    y1 = wmma16x16x32(Ap, Bg1, y1);   // y[q, i=16..31]

    __syncthreads();
  }

  #pragma unroll
  for (int r = 0; r < 8; ++r) {
    const int qsrc = r + (hi ? 8 : 0);
    const float linv = 1.f / __shfl(l_run, qsrc, 32);
    const int qg = q0 + wave * 16 + qsrc;
    YW[((size_t)(b * NPIX + qg)) * INTER + l15]      = y0[r] * linv;
    YW[((size_t)(b * NPIX + qg)) * INTER + 16 + l15] = y1[r] * linv;
  }
}

// ---------------- Kernel 5: out = Ww @ y + x -------------------------------------------
__global__ void __launch_bounds__(256) zout_kernel(const float* __restrict__ YW,
    const float* __restrict__ Ww, const float* __restrict__ x, float* __restrict__ out) {
  __shared__ float WwS[C1 * INTER];
  __shared__ float Ys[64 * INTER];
  const int b = blockIdx.y;
  const int n0 = blockIdx.x * 64;
  const int tid = threadIdx.x;
  for (int idx = tid; idx < C1 * INTER; idx += 256) WwS[idx] = Ww[idx];
  for (int idx = tid; idx < 64 * INTER; idx += 256)
    Ys[idx] = YW[((size_t)(b * NPIX) + n0) * INTER + idx];
  __syncthreads();
  const int nl = tid & 63;
  const int n = n0 + nl;
  float yr[INTER];
  #pragma unroll
  for (int i = 0; i < INTER; ++i) yr[i] = Ys[nl * INTER + i];
  for (int c = tid >> 6; c < C1; c += 4) {
    float acc = x[((size_t)(b * C1 + c)) * NPIX + n];
    #pragma unroll
    for (int i = 0; i < INTER; ++i) acc += WwS[c * INTER + i] * yr[i];
    out[((size_t)(b * C1 + c)) * NPIX + n] = acc;
  }
}

extern "C" void kernel_launch(void* const* d_in, const int* in_sizes, int n_in,
                              void* d_out, int out_size, void* d_ws, size_t ws_size,
                              hipStream_t stream) {
  const float* x  = (const float*)d_in[0];
  const float* Wg = (const float*)d_in[1];
  const float* Ww = (const float*)d_in[2];
  const float* W1 = (const float*)d_in[3];
  const float* W2 = (const float*)d_in[4];
  float* out = (float*)d_out;
  char* ws = (char*)d_ws;

  const size_t OFF_X3   = 0;
  const size_t OFF_X5   = OFF_X3 + (size_t)BATCH * C1 * NPIX * 4;
  const size_t OFF_PART = OFF_X5 + (size_t)BATCH * C1 * NPIX * 4;
  const size_t OFF_VV   = OFF_PART + (size_t)BATCH * C1 * 3 * 16 * 4;
  const size_t OFF_XT   = OFF_VV + 1024;
  const size_t OFF_GT   = OFF_XT + (size_t)BATCH * NPIX * C3 * 2;
  const size_t OFF_YW   = OFF_GT + (size_t)BATCH * INTER * NPIX * 2;
  const size_t NEED     = OFF_YW + (size_t)BATCH * NPIX * INTER * 4;
  if (ws_size < NEED) return;

  float*    X3   = (float*)(ws + OFF_X3);
  float*    X5   = (float*)(ws + OFF_X5);
  float*    PART = (float*)(ws + OFF_PART);
  float*    VV   = (float*)(ws + OFF_VV);
  _Float16* XTp  = (_Float16*)(ws + OFF_XT);
  _Float16* GTp  = (_Float16*)(ws + OFF_GT);
  float*    YW   = (float*)(ws + OFF_YW);

  pool_stats_kernel<<<dim3(16, C1, BATCH), 256, 0, stream>>>(x, X3, X5, PART);
  stats_kernel<<<dim3(BATCH), 256, 0, stream>>>(PART, W1, W2, VV);
  build_xt_kernel<<<dim3(64, 12, BATCH), 256, 0, stream>>>(x, X3, X5, VV, XTp);
  build_g_kernel<<<dim3(16, INTER, BATCH), 256, 0, stream>>>(x, Wg, GTp);
  flash_kernel<<<dim3(64, BATCH), 128, 0, stream>>>(XTp, GTp, YW);
  zout_kernel<<<dim3(64, BATCH), 256, 0, stream>>>(YW, Ww, x, out);
}
